// CfGCNController_31001073943308
// MI455X (gfx1250) — compile-verified, matrix operand order checked by reading, not run on hardware
//
#include <hip/hip_runtime.h>
#include <hip/hip_bf16.h>
#include <math.h>
#include <stdint.h>

typedef __bf16 bf16_t;
typedef __attribute__((ext_vector_type(16))) __bf16 v16bf;
typedef __attribute__((ext_vector_type(8)))  __bf16 v8bf;
typedef __attribute__((ext_vector_type(8)))  float  v8f;

#define HID   64
#define NN    64
#define CIN   128
#define ST    72      // padded stride for 64-wide bf16 LDS mats (144B = 9x16B chunks, coprime w/ 16 -> conflict-free)
#define EST   136     // padded stride for 128-wide bf16 LDS mats (272B = 17 chunks, conflict-free)
#define WST   64      // gate weights: unpadded stride + XOR chunk swizzle
#define BLOCK 512     // 16 wave32s == 16 output tiles of a 64x64 product

struct __align__(16) Smem {
    bf16_t Wg[18][HID * WST];  // gate weights, col-major, 16B-chunk XOR-swizzled
    bf16_t encW[HID * EST];    // encoder weight, col-major [e][c]
    union {                    // feats lives in encoder phase; Yh* live phases 3-4
        bf16_t feats[NN * EST];
        struct { bf16_t Yh1[NN * ST]; bf16_t Yh2[NN * ST]; } y;
    } u;
    bf16_t At_rm[NN * ST], At_cm[NN * ST];
    bf16_t S2_rm[NN * ST];
    bf16_t h_rm [NN * ST], h_cm [NN * ST];
    bf16_t x_rm [NN * ST], x_cm [NN * ST];
    bf16_t Yx1[NN * ST], Yx2[NN * ST];
    float stgA[NN * NN];       // async-prefetched adjacency tile (raw f32)
    float stgF[CIN * NN];      // async-prefetched frame tile (raw f32)
    float pooled[HID];
    float z1[128];
    float z2[64];
};

// ---- async global->LDS copy of one 16B chunk (ASYNCcnt-tracked) ----
__device__ __forceinline__ void async_ld16(uint32_t lds_byte, uint32_t voff, const float* base) {
    asm volatile("global_load_async_to_lds_b128 %0, %1, %2"
                 :: "v"(lds_byte), "v"(voff), "s"((uint64_t)(uintptr_t)base)
                 : "memory");
}
__device__ __forceinline__ void wait_async0() {
    asm volatile("s_wait_asynccnt 0x0" ::: "memory");
}
__device__ __forceinline__ uint32_t lds_addr32(const void* p) {
    return (uint32_t)(uintptr_t)p;   // flat LDS aperture: addr[31:0] is the LDS offset
}

// Load a 16x32 bf16 WMMA fragment (A from row-major store, B from col-major store).
__device__ __forceinline__ v16bf load_frag(const bf16_t* base, int stride, int i0, int k0, int lane) {
    int i  = i0 + (lane & 15);
    int h8 = (lane >> 4) << 3;
    const bf16_t* p = base + i * stride + k0 + h8;
    v8bf lo = *(const v8bf*)p;          // K = k0+h8    .. +7
    v8bf hi = *(const v8bf*)(p + 16);   // K = k0+16+h8 .. +7
    return __builtin_shufflevector(lo, hi, 0,1,2,3,4,5,6,7,8,9,10,11,12,13,14,15);
}

// Swizzled B-fragment load from a gate-weight matrix (stride 64, chunk^((row>>1)&7))
__device__ __forceinline__ v16bf load_frag_w(const bf16_t* base, int n0, int k0, int lane) {
    int i  = n0 + (lane & 15);
    int h8 = (lane >> 4) << 3;
    int sw = (i >> 1) & 7;
    int c0 = (k0 + h8) >> 3;            // logical 16B chunk within row
    const bf16_t* row = base + i * WST;
    v8bf lo = *(const v8bf*)(row + (((c0    ) ^ sw) << 3));
    v8bf hi = *(const v8bf*)(row + (((c0 + 2) ^ sw) << 3));
    return __builtin_shufflevector(lo, hi, 0,1,2,3,4,5,6,7,8,9,10,11,12,13,14,15);
}

// One 16x16 output tile of C(64xN) = A(64xK) @ B(KxN)
__device__ __forceinline__ v8f mm_tile(const bf16_t* Arm, int sA, const bf16_t* Bcm, int sB,
                                       int m0, int n0, int K, v8f acc, int lane) {
    #pragma unroll
    for (int k0 = 0; k0 < 128; k0 += 32) {
        if (k0 >= K) break;
        v16bf a = load_frag(Arm, sA, m0, k0, lane);
        v16bf b = load_frag(Bcm, sB, n0, k0, lane);
        acc = __builtin_amdgcn_wmma_f32_16x16x32_bf16(false, a, false, b, (short)0, acc, false, false);
    }
    return acc;
}
__device__ __forceinline__ v8f mm_tile_w(const bf16_t* Arm, const bf16_t* W,
                                         int m0, int n0, v8f acc, int lane) {
    #pragma unroll
    for (int k0 = 0; k0 < 64; k0 += 32) {
        v16bf a = load_frag(Arm, ST, m0, k0, lane);
        v16bf b = load_frag_w(W, n0, k0, lane);
        acc = __builtin_amdgcn_wmma_f32_16x16x32_bf16(false, a, false, b, (short)0, acc, false, false);
    }
    return acc;
}

// D-tile (f32): lane holds column n=lane&15, rows m0+8*(lane>>4)+r, r=0..7
__device__ __forceinline__ void store_cm(bf16_t* cm, int stride, int n0, int m0, v8f d, int lane) {
    int n  = n0 + (lane & 15);
    int mh = m0 + ((lane >> 4) << 3);
    v8bf v;
    #pragma unroll
    for (int r = 0; r < 8; ++r) v[r] = (__bf16)d[r];
    *(v8bf*)(cm + n * stride + mh) = v;
}
__device__ __forceinline__ void store_rm(bf16_t* rm, int stride, int m0, int n0, v8f d, int lane) {
    int n  = n0 + (lane & 15);
    int mh = m0 + ((lane >> 4) << 3);
    #pragma unroll
    for (int r = 0; r < 8; ++r) rm[(mh + r) * stride + n] = (__bf16)d[r];
}

__global__ __launch_bounds__(BLOCK)
void gclstm_kernel(const float* __restrict__ frames, const float* __restrict__ adj,
                   const float* __restrict__ hid0,
                   const float* __restrict__ encW, const float* __restrict__ encB,
                   const float* __restrict__ Wf_h, const float* __restrict__ Wf_u, const float* __restrict__ bf_,
                   const float* __restrict__ Wg_h, const float* __restrict__ Wg_u, const float* __restrict__ bg_,
                   const float* __restrict__ Wc_h, const float* __restrict__ Wc_u, const float* __restrict__ bc_,
                   const float* __restrict__ dW1, const float* __restrict__ db1,
                   const float* __restrict__ dW2, const float* __restrict__ db2,
                   const float* __restrict__ dW3, const float* __restrict__ db3,
                   const float* __restrict__ osc, const float* __restrict__ obi,
                   float* __restrict__ out, int B, int T) {
    __shared__ Smem sm;
    const int tid  = threadIdx.x;
    const int lane = tid & 31;
    const int wave = tid >> 5;
    const int m0 = (wave & 3) * 16;   // output tile row
    const int n0 = (wave >> 2) * 16;  // output tile col
    const int b  = blockIdx.x;
    const size_t hid_off = (size_t)B * T * 6;

    const uint32_t stgA_lds = lds_addr32(sm.stgA);
    const uint32_t stgF_lds = lds_addr32(sm.stgF);

    // ---- kick off async prefetch of t=0 tiles (overlaps weight preload) ----
    {
        const float* Ag = adj    + (size_t)b * T * (size_t)(NN * NN);
        const float* Fg = frames + (size_t)b * T * (size_t)(CIN * NN);
        #pragma unroll
        for (int i = 0; i < 2; ++i) {               // 1024 chunks of 16B
            uint32_t c = tid + i * BLOCK;
            async_ld16(stgA_lds + c * 16, c * 16, Ag);
        }
        #pragma unroll
        for (int i = 0; i < 4; ++i) {               // 2048 chunks of 16B
            uint32_t c = tid + i * BLOCK;
            async_ld16(stgF_lds + c * 16, c * 16, Fg);
        }
    }

    // ---- hoist per-thread loop-invariant scalars (keep them in VGPRs) ----
    const int  ncol = n0 + (lane & 15);
    const float eb = encB[ncol];
    const float bF = bf_[ncol], bG = bg_[ncol], bC = bc_[ncol];
    float r_db1 = 0.f, r_db2 = 0.f, r_db3 = 0.f, r_osc = 0.f, r_obi = 0.f;
    if (tid < 128) r_db1 = db1[tid];
    if (tid < 64)  r_db2 = db2[tid];
    if (tid < 6) { r_db3 = db3[tid]; r_osc = osc[tid]; r_obi = obi[tid]; }

    // ---- one-time preload: gate weights (col-major, swizzled bf16) ----
    const float* gsrc[18] = {
        Wf_h, Wf_h + 4096, Wf_h + 8192, Wf_u, Wf_u + 4096, Wf_u + 8192,
        Wg_h, Wg_h + 4096, Wg_h + 8192, Wg_u, Wg_u + 4096, Wg_u + 8192,
        Wc_h, Wc_h + 4096, Wc_h + 8192, Wc_u, Wc_u + 4096, Wc_u + 8192 };
    for (int idx = tid; idx < 18 * 4096; idx += BLOCK) {
        int w = idx >> 12, r = idx & 4095, d = r >> 6, e = r & 63;
        int phys = ((((d >> 3) ^ ((e >> 1) & 7)) << 3) | (d & 7));
        sm.Wg[w][e * WST + phys] = (bf16_t)gsrc[w][r];
    }
    for (int idx = tid; idx < CIN * HID; idx += BLOCK) {           // encW cm: [e][c]
        int c = idx >> 6, e = idx & 63;
        sm.encW[e * EST + c] = (bf16_t)encW[idx];
    }
    for (int idx = tid; idx < 4096; idx += BLOCK) {                // h0 rm + cm
        int m = idx >> 6, n = idx & 63;
        float v = hid0[(size_t)b * 4096 + idx];
        sm.h_rm[m * ST + n] = (bf16_t)v;
        sm.h_cm[n * ST + m] = (bf16_t)v;
    }
    if (tid < HID) sm.pooled[tid] = 0.f;
    __syncthreads();

    for (int t = 0; t < T; ++t) {
        // ---- phase 1: consume prefetched tiles, convert to bf16 layouts ----
        wait_async0();
        __syncthreads();                 // all waves' async chunks have landed
        for (int idx = tid; idx < 4096; idx += BLOCK) {
            int m = idx >> 6, n = idx & 63;
            float v = sm.stgA[idx];
            sm.At_rm[m * ST + n] = (bf16_t)v;
            sm.At_cm[n * ST + m] = (bf16_t)v;
        }
        for (int idx = tid; idx < CIN * NN; idx += BLOCK) {        // feats[n][c] = frames[c][n]
            int c = idx >> 6, n = idx & 63;
            sm.u.feats[n * EST + c] = (bf16_t)sm.stgF[idx];
        }
        __syncthreads();                 // staging consumed: safe to overwrite

        // ---- issue async prefetch of t+1 tiles; overlaps all WMMA phases ----
        if (t + 1 < T) {
            const float* Ag = adj    + ((size_t)b * T + t + 1) * (size_t)(NN * NN);
            const float* Fg = frames + ((size_t)b * T + t + 1) * (size_t)(CIN * NN);
            #pragma unroll
            for (int i = 0; i < 2; ++i) {
                uint32_t c = tid + i * BLOCK;
                async_ld16(stgA_lds + c * 16, c * 16, Ag);
            }
            #pragma unroll
            for (int i = 0; i < 4; ++i) {
                uint32_t c = tid + i * BLOCK;
                async_ld16(stgF_lds + c * 16, c * 16, Fg);
            }
        }

        // ---- phase 2: x = feats @ encW + b ; S2 = At @ At ----
        {
            v8f xa = {};
            xa = mm_tile(sm.u.feats, EST, sm.encW, EST, m0, n0, 128, xa, lane);
            #pragma unroll
            for (int r = 0; r < 8; ++r) xa[r] += eb;
            store_rm(sm.x_rm, ST, m0, n0, xa, lane);
            store_cm(sm.x_cm, ST, n0, m0, xa, lane);
            v8f s2 = {};
            s2 = mm_tile(sm.At_rm, ST, sm.At_cm, ST, m0, n0, 64, s2, lane);
            store_rm(sm.S2_rm, ST, m0, n0, s2, lane);
        }
        __syncthreads();                 // feats dead after this point

        // ---- phase 3: Y = {At,S2} @ {h,x}  (shared across the 3 gates) ----
        {
            v8f y;
            y = (v8f){}; y = mm_tile(sm.At_rm, ST, sm.h_cm, ST, m0, n0, 64, y, lane); store_rm(sm.u.y.Yh1, ST, m0, n0, y, lane);
            y = (v8f){}; y = mm_tile(sm.S2_rm, ST, sm.h_cm, ST, m0, n0, 64, y, lane); store_rm(sm.u.y.Yh2, ST, m0, n0, y, lane);
            y = (v8f){}; y = mm_tile(sm.At_rm, ST, sm.x_cm, ST, m0, n0, 64, y, lane); store_rm(sm.Yx1, ST, m0, n0, y, lane);
            y = (v8f){}; y = mm_tile(sm.S2_rm, ST, sm.x_cm, ST, m0, n0, 64, y, lane); store_rm(sm.Yx2, ST, m0, n0, y, lane);
        }
        __syncthreads();

        // ---- phase 4: gates = sum_j Aops[j] @ W[j] ----
        const bf16_t* Aops[6] = { sm.h_rm, sm.u.y.Yh1, sm.u.y.Yh2, sm.x_rm, sm.Yx1, sm.Yx2 };
        v8f af = {}, ag = {}, ac = {};
        #pragma unroll
        for (int j = 0; j < 6; ++j) {
            af = mm_tile_w(Aops[j], sm.Wg[0 * 6 + j], m0, n0, af, lane);
            ag = mm_tile_w(Aops[j], sm.Wg[1 * 6 + j], m0, n0, ag, lane);
            ac = mm_tile_w(Aops[j], sm.Wg[2 * 6 + j], m0, n0, ac, lane);
        }
        __syncthreads();   // everyone done reading old h_rm / Y buffers

        // ---- phase 5: nonlinearity, h update, pooled, (final h) ----
        v8f hn;
        #pragma unroll
        for (int r = 0; r < 8; ++r) {
            float f = af[r] + bF, g = ag[r] + bG, c = ac[r] + bC;
            float s = 1.f / (1.f + __expf(-f));
            hn[r] = s * tanhf(g) + (1.f - s) * tanhf(c);
        }
        store_rm(sm.h_rm, ST, m0, n0, hn, lane);
        store_cm(sm.h_cm, ST, n0, m0, hn, lane);
        {
            float psum = 0.f;
            #pragma unroll
            for (int r = 0; r < 8; ++r) psum += hn[r];
            atomicAdd(&sm.pooled[ncol], psum);   // ds_add_f32
        }
        if (t == T - 1) {
            int mh = m0 + ((lane >> 4) << 3);
            float* ho = out + hid_off + (size_t)b * 4096;
            #pragma unroll
            for (int r = 0; r < 8; ++r) ho[(mh + r) * 64 + ncol] = hn[r];
        }
        __syncthreads();

        // ---- phase 6: decoder MLP on pooled mean ----
        if (tid < 128) {
            float a = r_db1;
            #pragma unroll 8
            for (int i = 0; i < 64; ++i) a += (sm.pooled[i] * 0.015625f) * dW1[i * 128 + tid];
            sm.z1[tid] = fmaxf(a, 0.f);
        }
        __syncthreads();
        if (tid < 64) {
            float a = r_db2;
            #pragma unroll 8
            for (int i = 0; i < 128; ++i) a += sm.z1[i] * dW2[i * 64 + tid];
            sm.z2[tid] = fmaxf(a, 0.f);
        } else if (tid < 128) {
            sm.pooled[tid - 64] = 0.f;           // reset for next step
        }
        __syncthreads();
        if (tid < 6) {
            float a = r_db3;
            #pragma unroll 8
            for (int i = 0; i < 64; ++i) a += sm.z2[i] * dW3[i * 6 + tid];
            out[((size_t)b * T + t) * 6 + tid] = a * r_osc + r_obi;
        }
        __syncthreads();
    }
}

extern "C" void kernel_launch(void* const* d_in, const int* in_sizes, int n_in,
                              void* d_out, int out_size, void* d_ws, size_t ws_size,
                              hipStream_t stream) {
    const float* frames = (const float*)d_in[0];
    const float* adj    = (const float*)d_in[1];
    const float* hid0   = (const float*)d_in[2];
    const float* encW   = (const float*)d_in[3];
    const float* encB   = (const float*)d_in[4];
    const float* Wf_h   = (const float*)d_in[5];
    const float* Wf_u   = (const float*)d_in[6];
    const float* bf_    = (const float*)d_in[7];
    const float* Wg_h   = (const float*)d_in[8];
    const float* Wg_u   = (const float*)d_in[9];
    const float* bg_    = (const float*)d_in[10];
    const float* Wc_h   = (const float*)d_in[11];
    const float* Wc_u   = (const float*)d_in[12];
    const float* bc_    = (const float*)d_in[13];
    const float* dW1    = (const float*)d_in[14];
    const float* db1    = (const float*)d_in[15];
    const float* dW2    = (const float*)d_in[16];
    const float* db2    = (const float*)d_in[17];
    const float* dW3    = (const float*)d_in[18];
    const float* db3    = (const float*)d_in[19];
    const float* osc    = (const float*)d_in[20];
    const float* obi    = (const float*)d_in[21];

    int B = in_sizes[2] / 4096;            // hidden_state: B*1*64*64
    int T = in_sizes[1] / (B * 4096);      // adjacency:    B*T*64*64

    gclstm_kernel<<<dim3(B), dim3(BLOCK), 0, stream>>>(
        frames, adj, hid0, encW, encB,
        Wf_h, Wf_u, bf_, Wg_h, Wg_u, bg_, Wc_h, Wc_u, bc_,
        dW1, db1, dW2, db2, dW3, db3, osc, obi,
        (float*)d_out, B, T);
}
